// EnhancedCTRGCN_32418413150864
// MI455X (gfx1250) — matrix operand order repaired and checked by reading, not compiled, not run on hardware
//
#include <hip/hip_runtime.h>

typedef _Float16 half_t;
typedef __attribute__((ext_vector_type(16))) _Float16 v16h;
typedef __attribute__((ext_vector_type(8)))  _Float16 v8h;
typedef __attribute__((ext_vector_type(8)))  float    v8f;

#define NBATCH 64
#define NV 25
#define T0 256
#define ALPHA_C 0.85f
#define EPS_C 1e-5f
#define KP 4

// ---------------- small utility kernels ----------------

__global__ void k_zero(float* p, int n) {
  int i = blockIdx.x * blockDim.x + threadIdx.x;
  if (i < n) p[i] = 0.f;
}

__global__ void k_cvt(const float* __restrict__ w, half_t* __restrict__ o, int n) {
  int i = blockIdx.x * blockDim.x + threadIdx.x;
  if (i < n) o[i] = (half_t)w[i];
}

// tcn_w (co, ci, 9, 1) -> w2[o][k*CI + c]  (K-order = tap outer, channel inner)
__global__ void k_tcnw(const float* __restrict__ w, half_t* __restrict__ o2, int CI, int CO) {
  int i = blockIdx.x * blockDim.x + threadIdx.x;
  int tot = CO * CI * 9;
  if (i >= tot) return;
  int k = i % 9;
  int c = (i / 9) % CI;
  int o = i / (9 * CI);
  o2[(size_t)o * 9 * CI + (size_t)k * CI + c] = (half_t)w[i];
}

// ---------------- data_bn (channel = c*V + v, reduce over N,T) ----------------

__global__ void k_bn0_stats(const float* __restrict__ x, const float* __restrict__ g,
                            const float* __restrict__ b, float* __restrict__ scale,
                            float* __restrict__ shift) {
  int ch = blockIdx.x;           // 0..49
  int c = ch / NV, v = ch % NV;
  float s = 0.f, s2 = 0.f;
  for (int j = threadIdx.x; j < NBATCH * T0; j += blockDim.x) {
    int n = j / T0, t = j % T0;
    float val = x[(((size_t)n * 2 + c) * T0 + t) * NV + v];
    s += val; s2 += val * val;
  }
  __shared__ float sh[256], sh2[256];
  sh[threadIdx.x] = s; sh2[threadIdx.x] = s2;
  __syncthreads();
  for (int w = 128; w > 0; w >>= 1) {
    if (threadIdx.x < w) { sh[threadIdx.x] += sh[threadIdx.x + w]; sh2[threadIdx.x] += sh2[threadIdx.x + w]; }
    __syncthreads();
  }
  if (threadIdx.x == 0) {
    float inv = 1.f / (float)(NBATCH * T0);
    float m = sh[0] * inv;
    float var = sh2[0] * inv - m * m;
    float sc = g[ch] * rsqrtf(var + EPS_C);
    scale[ch] = sc; shift[ch] = b[ch] - m * sc;
  }
}

// normalize and transpose (N,C,T,V) -> f16 [n][t][v][c]
__global__ void k_bn0_apply(const float* __restrict__ x, const float* __restrict__ scale,
                            const float* __restrict__ shift, half_t* __restrict__ out) {
  int i = blockIdx.x * blockDim.x + threadIdx.x;
  if (i >= NBATCH * 2 * T0 * NV) return;
  int v = i % NV;
  int t = (i / NV) % T0;
  int c = (i / (NV * T0)) % 2;
  int n = i / (NV * T0 * 2);
  int ch = c * NV + v;
  out[(((size_t)n * T0 + t) * NV + v) * 2 + c] = (half_t)(x[i] * scale[ch] + shift[ch]);
}

// ---------------- graph conv: y[nt][w][c] = sum_v (A0+Ares)[w][v] * x[nt][v][c] ----------------

__global__ void k_graph(const half_t* __restrict__ act, const float* __restrict__ A0,
                        const float* __restrict__ Ares, half_t* __restrict__ y, int C) {
  __shared__ float As[NV * NV];
  __shared__ float Xs[NV * 128];
  size_t nt = blockIdx.x;
  const half_t* src = act + nt * NV * C;
  for (int i = threadIdx.x; i < NV * NV; i += blockDim.x) As[i] = A0[i] + Ares[i];
  for (int i = threadIdx.x; i < NV * C; i += blockDim.x) Xs[i] = (float)src[i];
  __syncthreads();
  for (int i = threadIdx.x; i < NV * C; i += blockDim.x) {
    int w = i / C, c = i % C;
    float s = 0.f;
#pragma unroll
    for (int v = 0; v < NV; ++v) s += As[w * NV + v] * Xs[v * C + c];
    y[nt * NV * C + i] = (half_t)s;
  }
}

// ---------------- tiny-K (ci=2) channel matmul (block 1 fc / residual) ----------------

__global__ void k_lin2(const half_t* __restrict__ act, const float* __restrict__ w,
                       const float* __restrict__ bias, float* __restrict__ out,
                       int M, int CO) {
  int i = blockIdx.x * blockDim.x + threadIdx.x;
  if (i >= M * CO) return;
  int o = i % CO;
  size_t m = (size_t)(i / CO);
  float s = (float)act[m * 2] * w[o * 2] + (float)act[m * 2 + 1] * w[o * 2 + 1];
  if (bias) s += bias[o];
  out[i] = s;
}

// ---------------- BN statistics (sum / sumsq per channel, float atomics) ----------------

__global__ void k_stats(const float* __restrict__ z, float* __restrict__ sums,
                        int M, int C, int rpb) {
  int o = threadIdx.x % C;
  int sub = threadIdx.x / C;
  int nsub = blockDim.x / C;
  long m0 = (long)blockIdx.x * rpb;
  long mend = m0 + rpb; if (mend > M) mend = M;
  float s = 0.f, s2 = 0.f;
  for (long m = m0 + sub; m < mend; m += nsub) {
    float v = z[m * C + o];
    s += v; s2 += v * v;
  }
  atomicAdd(&sums[o], s);
  atomicAdd(&sums[C + o], s2);
}

__global__ void k_bnfin(const float* __restrict__ sums, const float* __restrict__ g,
                        const float* __restrict__ b, float* __restrict__ scale,
                        float* __restrict__ shift, int C, float invM) {
  int o = blockIdx.x * blockDim.x + threadIdx.x;
  if (o >= C) return;
  float m = sums[o] * invM;
  float var = sums[C + o] * invM - m * m;
  float sc = g[o] * rsqrtf(var + EPS_C);
  scale[o] = sc;
  shift[o] = b[o] - m * sc;
}

// BN + ReLU -> f16 (feeds the temporal-conv WMMA GEMM)
__global__ void k_bnrelu(const float* __restrict__ z, const float* __restrict__ scale,
                         const float* __restrict__ shift, half_t* __restrict__ h,
                         long total, int C) {
  long i = (long)blockIdx.x * blockDim.x + threadIdx.x;
  if (i >= total) return;
  int o = (int)(i % C);
  h[i] = (half_t)fmaxf(z[i] * scale[o] + shift[o], 0.f);
}

// ---------------- WMMA GEMM: out[m][o] = sum_k act[(n, tout*stride, v)][k] * wt[o][k] ----------------
// act: [NB][Tin][V][K] f16 (K contiguous), wt: [CO][K] f16 row-major, out: [NB][Tout][V][CO] f32
// One wave per 16x16 tile; K multiple of 32. Used for fc (stride=1) and residual (stride=s).

__global__ __launch_bounds__(256) void k_gemm(
    const half_t* __restrict__ act, const half_t* __restrict__ wt,
    const float* __restrict__ bias, float* __restrict__ out,
    int Tin, int Tout, int tstride, int K, int CO) {
  int wave = threadIdx.x >> 5, lane = threadIdx.x & 31;
  int hlf = lane >> 4, l15 = lane & 15;
  long Mtot = (long)NBATCH * Tout * NV;
  long mtile = (long)blockIdx.x * 8 + wave;
  if (mtile * 16 >= Mtot) return;          // wave-uniform exit (EXEC all-1 for WMMA)
  int otile = blockIdx.y;

  long m = mtile * 16 + l15;               // per-lane output row (A-matrix row)
  int v = (int)(m % NV);
  long nt = m / NV;
  int tout = (int)(nt % Tout);
  int n = (int)(nt / Tout);
  const half_t* arow = act + ((((size_t)n * Tin + (size_t)tout * tstride) * NV + v) * K);
  const half_t* wrow = wt + (size_t)(otile * 16 + l15) * K;

  v8f acc = {};
  for (int k0 = 0; k0 < K; k0 += 32) {
    v16h a, bb;
    // A (16-bit 16x32): lane half h holds K in {16h..16h+15}
    a = *(const v16h*)(arow + k0 + 16 * hlf);
    // B (16-bit 32x16): lane half h holds K in {16h..16h+15} for column l15
    bb = *(const v16h*)(wrow + k0 + 16 * hlf);
    acc = __builtin_amdgcn_wmma_f32_16x16x32_f16(false, a, false, bb, (short)0, acc,
                                                 false, false);
  }
  long mbase = mtile * 16 + hlf * 8;
  int oc = otile * 16 + l15;
  float bv = bias ? bias[oc] : 0.f;
#pragma unroll
  for (int r = 0; r < 8; ++r)
    out[(mbase + r) * CO + oc] = acc[r] + bv;   // D: vgpr r = row r+8*hlf, lane l15 = col
}

// ---------------- temporal 9x1 conv as implicit WMMA GEMM (K = 9*CI) ----------------

__global__ __launch_bounds__(256) void k_tcn(
    const half_t* __restrict__ h, const half_t* __restrict__ wt,
    float* __restrict__ out, int Tin, int Tout, int tstride, int CI, int CO) {
  int wave = threadIdx.x >> 5, lane = threadIdx.x & 31;
  int hlf = lane >> 4, l15 = lane & 15;
  long Mtot = (long)NBATCH * Tout * NV;
  long mtile = (long)blockIdx.x * 8 + wave;
  if (mtile * 16 >= Mtot) return;
  int otile = blockIdx.y;

  long m = mtile * 16 + l15;
  int v = (int)(m % NV);
  long nt = m / NV;
  int tout = (int)(nt % Tout);
  int n = (int)(nt / Tout);
  const half_t* wrow = wt + (size_t)(otile * 16 + l15) * (9 * CI);

  v8f acc = {};
  int Ktot = 9 * CI;
  for (int k0 = 0; k0 < Ktot; k0 += 32) {    // CI%32==0 -> tile lies inside one tap
    int tap = k0 / CI;
    int cb = k0 - tap * CI;
    int tin = tout * tstride + tap - KP;
    v16h a = {};
    if (tin >= 0 && tin < Tin) {
      const half_t* pa = h + ((((size_t)n * Tin + tin) * NV + v) * CI + cb);
      a = *(const v16h*)(pa + 16 * hlf);
      __builtin_prefetch(pa + (size_t)NV * CI, 0, 0);   // next tap's row -> global_prefetch
    }
    v16h bb = *(const v16h*)(wrow + k0 + 16 * hlf);
    acc = __builtin_amdgcn_wmma_f32_16x16x32_f16(false, a, false, bb, (short)0, acc,
                                                 false, false);
  }
  long mbase = mtile * 16 + hlf * 8;
  int oc = otile * 16 + l15;
#pragma unroll
  for (int r = 0; r < 8; ++r)
    out[(mbase + r) * CO + oc] = acc[r];     // tcn bias cancels in following BN
}

// ---------------- BN2 + residual + ReLU (in place on u) ----------------

__global__ void k_combine(float* __restrict__ u, const float* __restrict__ res,
                          const float* __restrict__ scale, const float* __restrict__ shift,
                          long total, int C) {
  long i = (long)blockIdx.x * blockDim.x + threadIdx.x;
  if (i >= total) return;
  int o = (int)(i % C);
  u[i] = fmaxf(u[i] * scale[o] + shift[o] + res[i], 0.f);
}

// ---------------- temporal EMA (register carry per (n,v,c)) -> f16 next-block input ----------------

__global__ void k_smooth(const float* __restrict__ u, half_t* __restrict__ act, int Tt, int C) {
  int i = blockIdx.x * blockDim.x + threadIdx.x;
  if (i >= NBATCH * NV * C) return;
  int o = i % C;
  int vv = (i / C) % NV;
  int n = i / (C * NV);
  size_t base = ((size_t)n * Tt) * NV * C + (size_t)vv * C + o;
  size_t str = (size_t)NV * C;
  float carry = u[base];
  act[base] = (half_t)carry;
  for (int t = 1; t < Tt; ++t) {
    carry = ALPHA_C * carry + (1.f - ALPHA_C) * u[base + (size_t)t * str];
    act[base + (size_t)t * str] = (half_t)carry;
  }
}

// ---------------- global average pool + classifier head ----------------

__global__ void k_pool(const half_t* __restrict__ act, float* __restrict__ pooled, int Tt, int C) {
  int n = blockIdx.x;
  for (int o = threadIdx.x; o < C; o += blockDim.x) {
    float s = 0.f;
    for (int tv = 0; tv < Tt * NV; ++tv)
      s += (float)act[((size_t)n * Tt * NV + tv) * C + o];
    pooled[n * C + o] = s / (float)(Tt * NV);
  }
}

__global__ void k_head(const float* __restrict__ pooled, const float* __restrict__ fw,
                       const float* __restrict__ fb, float* __restrict__ out) {
  int n = blockIdx.x, j = threadIdx.x;
  if (j >= 60) return;
  float s = fb[j];
  for (int o = 0; o < 256; ++o) s += pooled[n * 256 + o] * fw[j * 256 + o];
  out[n * 60 + j] = s;
}

// ---------------- host-side orchestration ----------------

extern "C" void kernel_launch(void* const* d_in, const int* in_sizes, int n_in,
                              void* d_out, int out_size, void* d_ws, size_t ws_size,
                              hipStream_t stream) {
  (void)in_sizes; (void)n_in; (void)out_size; (void)ws_size;

  const float* x     = (const float*)d_in[0];
  const float* Ainit = (const float*)d_in[1];
  const float* bn0g  = (const float*)d_in[2];
  const float* bn0b  = (const float*)d_in[3];
  auto BP = [&](int b, int k) -> const float* { return (const float*)d_in[4 + 11 * b + k]; };
  // per-block: 0 A_res, 1 fc_w, 2 fc_b(unused: cancels in BN1), 3 bn1_g, 4 bn1_b,
  //            5 tcn_w, 6 tcn_b(unused: cancels in BN2), 7 bn2_g, 8 bn2_b, 9 res_w, 10 res_b
  const float* head_w = (const float*)d_in[37];
  const float* head_b = (const float*)d_in[38];

  char* ws = (char*)d_ws;
  float* small     = (float*)ws;
  float* bn0_scale = small + 0;      // 50
  float* bn0_shift = small + 64;     // 50
  float* sums      = small + 128;    // 512
  float* scl       = small + 640;    // 256
  float* shf       = small + 896;    // 256
  float* pooled    = small + 1152;   // 16384

  half_t* wbuf = (half_t*)(ws + (128 << 10));
  half_t* fcw2 = wbuf + 0;           // 128*64
  half_t* fcw3 = wbuf + 8192;        // 256*128
  half_t* rsw2 = wbuf + 40960;       // 128*64
  half_t* rsw3 = wbuf + 49152;       // 256*128
  half_t* tw1  = wbuf + 81920;       // 64*576
  half_t* tw2  = wbuf + 118784;      // 128*1152
  half_t* tw3  = wbuf + 266240;      // 256*2304

  char* big = ws + (128 << 10) + (4 << 20);
  half_t* actA = (half_t*)big;                                  // 26,214,400 halves
  half_t* yh   = (half_t*)(big + 52428800ull);                  // 52,428,800 halves (y then h16)
  float*  zf   = (float*)(big + 52428800ull + 104857600ull);    // 52,428,800 floats
  float*  uf   = zf;                                            // reuse after z dead
  float*  rf   = zf + 26214400ull;                              // residual

  // --- weight precompute (f16 / reordered) ---
  k_cvt <<<(8192  + 255) / 256, 256, 0, stream>>>(BP(1, 1), fcw2, 8192);
  k_cvt <<<(32768 + 255) / 256, 256, 0, stream>>>(BP(2, 1), fcw3, 32768);
  k_cvt <<<(8192  + 255) / 256, 256, 0, stream>>>(BP(1, 9), rsw2, 8192);
  k_cvt <<<(32768 + 255) / 256, 256, 0, stream>>>(BP(2, 9), rsw3, 32768);
  k_tcnw<<<(36864  + 255) / 256, 256, 0, stream>>>(BP(0, 5), tw1, 64, 64);
  k_tcnw<<<(147456 + 255) / 256, 256, 0, stream>>>(BP(1, 5), tw2, 128, 128);
  k_tcnw<<<(589824 + 255) / 256, 256, 0, stream>>>(BP(2, 5), tw3, 256, 256);

  // --- data_bn ---
  k_bn0_stats<<<50, 256, 0, stream>>>(x, bn0g, bn0b, bn0_scale, bn0_shift);
  k_bn0_apply<<<(819200 + 255) / 256, 256, 0, stream>>>(x, bn0_scale, bn0_shift, actA);

  struct Blk { int Tin, Tout, st, ci, co; };
  const Blk blks[3] = { {256, 256, 1, 2, 64}, {256, 128, 2, 64, 128}, {128, 64, 2, 128, 256} };
  half_t* fcws[3] = { nullptr, fcw2, fcw3 };
  half_t* rsws[3] = { nullptr, rsw2, rsw3 };
  half_t* tws[3]  = { tw1, tw2, tw3 };

  for (int b = 0; b < 3; ++b) {
    const Blk& B = blks[b];
    long Min  = (long)NBATCH * B.Tin * NV;
    long Mout = (long)NBATCH * B.Tout * NV;

    // graph conv over V
    k_graph<<<NBATCH * B.Tin, 256, 0, stream>>>(actA, Ainit, BP(b, 0), yh, B.ci);

    // fc (1x1 channel conv). fc_b dropped: cancelled by training-mode BN1.
    if (b == 0) {
      k_lin2<<<(int)((Min * B.co + 255) / 256), 256, 0, stream>>>(yh, BP(0, 1), nullptr, zf,
                                                                  (int)Min, B.co);
    } else {
      dim3 g((unsigned)((Min / 16 + 7) / 8), (unsigned)(B.co / 16));
      k_gemm<<<g, 256, 0, stream>>>(yh, fcws[b], nullptr, zf, B.Tin, B.Tin, 1, B.ci, B.co);
    }

    // BN1 stats + apply + ReLU -> f16 (into yh, y is dead)
    k_zero<<<2, 256, 0, stream>>>(sums, 512);
    k_stats<<<(int)((Min + 4095) / 4096), 256, 0, stream>>>(zf, sums, (int)Min, B.co, 4096);
    k_bnfin<<<1, 256, 0, stream>>>(sums, BP(b, 3), BP(b, 4), scl, shf, B.co, 1.f / (float)Min);
    k_bnrelu<<<(int)((Min * B.co + 255) / 256), 256, 0, stream>>>(zf, scl, shf, yh,
                                                                  Min * B.co, B.co);

    // temporal 9x1 conv (implicit WMMA GEMM). tcn_b dropped: cancelled by BN2.
    {
      dim3 g((unsigned)((Mout / 16 + 7) / 8), (unsigned)(B.co / 16));
      k_tcn<<<g, 256, 0, stream>>>(yh, tws[b], uf, B.Tin, B.Tout, B.st, B.co, B.co);
    }

    // BN2 stats
    k_zero<<<2, 256, 0, stream>>>(sums, 512);
    k_stats<<<(int)((Mout + 4095) / 4096), 256, 0, stream>>>(uf, sums, (int)Mout, B.co, 4096);
    k_bnfin<<<1, 256, 0, stream>>>(sums, BP(b, 7), BP(b, 8), scl, shf, B.co, 1.f / (float)Mout);

    // residual (strided 1x1, includes res_b)
    if (b == 0) {
      k_lin2<<<(int)((Mout * B.co + 255) / 256), 256, 0, stream>>>(actA, BP(0, 9), BP(0, 10),
                                                                   rf, (int)Mout, B.co);
    } else {
      dim3 g((unsigned)((Mout / 16 + 7) / 8), (unsigned)(B.co / 16));
      k_gemm<<<g, 256, 0, stream>>>(actA, rsws[b], BP(b, 10), rf, B.Tin, B.Tout, B.st,
                                    B.ci, B.co);
    }

    // BN2 + residual + ReLU (in place on u), then EMA -> next block input (f16)
    k_combine<<<(int)((Mout * B.co + 255) / 256), 256, 0, stream>>>(uf, rf, scl, shf,
                                                                    Mout * B.co, B.co);
    k_smooth<<<(NBATCH * NV * B.co + 255) / 256, 256, 0, stream>>>(uf, actA, B.Tout, B.co);
  }

  // pool over (t,v) and classifier
  k_pool<<<NBATCH, 256, 0, stream>>>(actA, pooled, 64, 256);
  k_head<<<NBATCH, 64, 0, stream>>>(pooled, head_w, head_b, (float*)d_out);
}
